// GraphAttentionNetwork_31662498906492
// MI455X (gfx1250) — compile-verified, matrix operand order checked by reading, not run on hardware
//
#include <hip/hip_runtime.h>
#include <hip/hip_bf16.h>
#include <math.h>

typedef __attribute__((ext_vector_type(16))) __bf16       v16bf;
typedef __attribute__((ext_vector_type(8)))  float        v8f;
typedef __attribute__((ext_vector_type(8)))  unsigned int v8u;
typedef __attribute__((ext_vector_type(4)))  unsigned int v4u;
typedef __attribute__((ext_vector_type(8)))  int          v8i;
typedef __attribute__((ext_vector_type(4)))  int          v4i;

#define N_NODES  20000
#define EA_EDGES 320000
#define EE_EDGES 200000
#define IN_DIM_C 96
#define EDGE_IN_C 197
#define EMB_C    8

// ---------------------------------------------------------------------------
// helpers
// ---------------------------------------------------------------------------
__device__ __forceinline__ void atomicMaxF(float* addr, float v) {
  // monotonic int/uint trick: correct ordered float atomic max
  if (v >= 0.f) atomicMax((int*)addr, __float_as_int(v));
  else          atomicMin((unsigned int*)addr, __float_as_uint(v));
}

// load a 16-element bf16 fragment: two 16-byte vectors at +0B and +32B
// (K-pair layout: VGPR0..3 = K {0..7}, VGPR4..7 = K {16..23}, per lane half)
__device__ __forceinline__ v16bf load_frag(const __bf16* base) {
  const uint4* p = (const uint4*)base;
  uint4 x0 = p[0], x1 = p[2];
  v8u u = {x0.x, x0.y, x0.z, x0.w, x1.x, x1.y, x1.z, x1.w};
  return __builtin_bit_cast(v16bf, u);
}

// ---------------------------------------------------------------------------
// conversion kernels
// ---------------------------------------------------------------------------
__global__ void k_f32_to_bf16(const float* __restrict__ in, __bf16* __restrict__ out, int n) {
  int i = blockIdx.x * blockDim.x + threadIdx.x;
  if (i < n) out[i] = (__bf16)in[i];
}

// W (K x Ncols, f32, row-major) -> Wt (Ncols x K, bf16, row-major)
__global__ void k_transpose_to_bf16(const float* __restrict__ W, __bf16* __restrict__ Wt,
                                    int K, int Ncols) {
  int i = blockIdx.x * blockDim.x + threadIdx.x;
  if (i < K * Ncols) {
    int n = i / K, k = i - n * K;
    Wt[i] = (__bf16)W[(size_t)k * Ncols + n];
  }
}

__global__ void k_fill(float* __restrict__ p, float v, int n) {
  int i = blockIdx.x * blockDim.x + threadIdx.x;
  if (i < n) p[i] = v;
}

// ---------------------------------------------------------------------------
// bf16 WMMA GEMM:  C[N x Ncols] = X[N x K] @ W[K x Ncols] + bias
// X bf16 row-major, W given as Wt = W^T (Ncols x K) bf16 row-major.
// Block = 256 threads (8 waves). blockIdx.x -> 16-row tile, staged into LDS
// by the Tensor Data Mover. Each wave owns TWO 16-col tiles (tiles_n is even
// for every layer: 32/16/2), so each A fragment feeds 2 WMMAs. K loop is
// software-pipelined: fragments for kk+32 load while the kk WMMAs issue.
// ---------------------------------------------------------------------------
__global__ void __launch_bounds__(256)
k_gemm_bf16_wmma(const __bf16* __restrict__ X, const __bf16* __restrict__ Wt,
                 const float* __restrict__ bias, float* __restrict__ C,
                 int K, int Ncols) {
  __shared__ __attribute__((aligned(16))) __bf16 xs[16 * 512];   // K <= 512 -> 16 KB

  const int row0 = blockIdx.x * 16;

  // ---- stage X tile (16 contiguous rows = one 16*K*2-byte chunk) into LDS ----
#if defined(__gfx1250__) && __has_builtin(__builtin_amdgcn_tensor_load_to_lds) && \
    __has_builtin(__builtin_amdgcn_s_wait_tensorcnt)
  if (threadIdx.x < 32) {                 // wave 0 issues the TDM op (EXEC ignored by TDM)
    const unsigned long long gaddr =
        (unsigned long long)(const void*)(X + (size_t)row0 * K);
    const unsigned ndw = (unsigned)(16 * K) >> 1;   // tile size in DWORDs (<= 4096)
    v4u g0; v8i g1; v4i gz4; v8i gz8;
    // D# group 0: count=1 | lds_addr (xs is the only LDS object -> offset 0)
    //             | global_addr[56:0] | type=2 ("image") in bits 127:126
    g0[0] = 1u;
    g0[1] = 0u;
    g0[2] = (unsigned)(gaddr & 0xFFFFFFFFull);
    g0[3] = (unsigned)((gaddr >> 32) & 0x01FFFFFFull) | 0x80000000u;
    // D# group 1: workgroup_mask=0, data_size=4B, 1-D tile: tile_dim0=ndw,
    //             tensor_dim0=ndw, tensor_dim1=1, stride0=ndw
    g1[0] = 2 << 16;                                   // data_size=4B
    g1[1] = (int)((ndw & 0xFFFFu) << 16);              // tensor_dim0[15:0]
    g1[2] = (int)((ndw >> 16) & 0xFFFFu) | (1 << 16);  // tensor_dim0[31:16] | tensor_dim1=1
    g1[3] = (int)(ndw << 16);                          // tile_dim0 = ndw
    g1[4] = 0;                                         // tile_dim1/2 unused
    g1[5] = (int)ndw;                                  // tensor_dim0_stride[31:0]
    g1[6] = 0; g1[7] = 0;
    gz4[0] = 0; gz4[1] = 0; gz4[2] = 0; gz4[3] = 0;
    gz8[0] = 0; gz8[1] = 0; gz8[2] = 0; gz8[3] = 0;
    gz8[4] = 0; gz8[5] = 0; gz8[6] = 0; gz8[7] = 0;
    __builtin_amdgcn_tensor_load_to_lds(g0, g1, gz4, gz4, gz8, 0);
    __builtin_amdgcn_s_wait_tensorcnt(0);
  }
#else
  {
    const unsigned int* Xu  = (const unsigned int*)(X + (size_t)row0 * K);
    unsigned int*       xsu = (unsigned int*)xs;
    const int nwords = (16 * K) >> 1;
    for (int i = threadIdx.x; i < nwords; i += 256) xsu[i] = Xu[i];
  }
#endif
  // Escape xs into an asm with a memory clobber: the TDM engine's LDS write is
  // invisible to alias analysis, and without an escaping use the LDS-global
  // pass proves "no stores" and folds every xs load to poison (observed: ds=0).
  {
    void* escape = (void*)xs;
    asm volatile("" : "+v"(escape) : : "memory");
  }
  __syncthreads();

  const int wave = threadIdx.x >> 5;
  const int lane = threadIdx.x & 31;
  const int hlf  = lane >> 4;              // lane half selects K sub-range
  const int r    = lane & 15;              // A-row / B-col index for this lane

  // readfirstlane -> provably wave-uniform guards: scalar branches, EXEC stays
  // all-ones around every WMMA (ISA requirement).
  const int  tbase = __builtin_amdgcn_readfirstlane((blockIdx.y * 8 + wave) * 2);
  if (tbase * 16 >= Ncols) return;
  const bool t1ok = (tbase + 1) * 16 < Ncols;   // uniform (tiles_n even => always true here)

  v8f acc0 = {}, acc1 = {};
  const __bf16* xrow = xs + r * K + 8 * hlf;
  const __bf16* w0   = Wt + (size_t)(tbase * 16 + r) * K + 8 * hlf;
  const __bf16* w1   = Wt + (size_t)((tbase + 1) * 16 + r) * K + 8 * hlf;

  // prologue: fragments for kk = 0
  v16bf a  = load_frag(xrow);
  v16bf b0 = load_frag(w0);
  v16bf b1 = a;
  if (t1ok) b1 = load_frag(w1);

  for (int kk = 32; kk < K; kk += 32) {
    // issue next iteration's loads first so they overlap the WMMAs below
    v16bf a_n  = load_frag(xrow + kk);
    v16bf b0_n = load_frag(w0 + kk);
    v16bf b1_n = a_n;
    if (t1ok) b1_n = load_frag(w1 + kk);
    if (kk + 32 < K) {
      __builtin_prefetch(w0 + kk + 32, 0, 1);          // global_prefetch_b8
      if (t1ok) __builtin_prefetch(w1 + kk + 32, 0, 1);
    }

    acc0 = __builtin_amdgcn_wmma_f32_16x16x32_bf16(false, a, false, b0, (short)0, acc0,
                                                   false, false);
    if (t1ok)
      acc1 = __builtin_amdgcn_wmma_f32_16x16x32_bf16(false, a, false, b1, (short)0, acc1,
                                                     false, false);
    a = a_n; b0 = b0_n; b1 = b1_n;
  }
  // epilogue: last K step
  acc0 = __builtin_amdgcn_wmma_f32_16x16x32_bf16(false, a, false, b0, (short)0, acc0,
                                                 false, false);
  if (t1ok)
    acc1 = __builtin_amdgcn_wmma_f32_16x16x32_bf16(false, a, false, b1, (short)0, acc1,
                                                   false, false);

  // C/D layout: VGPR i -> M = i + 8*half, lane&15 -> N
  {
    const float bv = bias[tbase * 16 + r];
#pragma unroll
    for (int i = 0; i < 8; ++i) {
      int m = row0 + i + 8 * hlf;
      C[(size_t)m * Ncols + tbase * 16 + r] = acc0[i] + bv;
    }
  }
  if (t1ok) {
    const float bv = bias[(tbase + 1) * 16 + r];
#pragma unroll
    for (int i = 0; i < 8; ++i) {
      int m = row0 + i + 8 * hlf;
      C[(size_t)m * Ncols + (tbase + 1) * 16 + r] = acc1[i] + bv;
    }
  }
}

// ---------------------------------------------------------------------------
// edge-attr embedding: emb[e] = attr[e] (197) @ W_edge (197x8)
// ---------------------------------------------------------------------------
__global__ void k_edge_embed(const float* __restrict__ attr, const float* __restrict__ Wedge,
                             float* __restrict__ emb, int ne) {
  int e = blockIdx.x * blockDim.x + threadIdx.x;
  if (e >= ne) return;
  const float* a = attr + (size_t)e * EDGE_IN_C;
  float acc[EMB_C];
#pragma unroll
  for (int j = 0; j < EMB_C; ++j) acc[j] = 0.f;
  for (int t = 0; t < EDGE_IN_C; ++t) {
    float av = a[t];
#pragma unroll
    for (int j = 0; j < EMB_C; ++j) acc[j] += av * Wedge[t * EMB_C + j];
  }
#pragma unroll
  for (int j = 0; j < EMB_C; ++j) emb[(size_t)e * EMB_C + j] = acc[j];
}

// ---------------------------------------------------------------------------
// pass 1: alpha = <q[dst], k[src] + e> * scale ; segment-max into amax
// ---------------------------------------------------------------------------
__global__ void k_attn_alpha(const float* __restrict__ q, const float* __restrict__ k,
                             const float* __restrict__ emb, const float* __restrict__ We,
                             const int* __restrict__ src, const int* __restrict__ dst,
                             float* __restrict__ alpha, float* __restrict__ amax,
                             int heads, int dout, int hc, int tot, float scale) {
  int idx = blockIdx.x * blockDim.x + threadIdx.x;
  if (idx >= tot) return;
  int e = idx / heads, h = idx - e * heads;
  int s = src[e], d = dst[e];
  const float* qr  = q + (size_t)d * hc + h * dout;
  const float* kr  = k + (size_t)s * hc + h * dout;
  const float* em  = emb + (size_t)e * EMB_C;
  const float* Wec = We + h * dout;
  float acc = 0.f;
  for (int c = 0; c < dout; ++c) {
    float ec = 0.f;
#pragma unroll
    for (int j = 0; j < EMB_C; ++j) ec += em[j] * Wec[(size_t)j * hc + c];
    acc += qr[c] * (kr[c] + ec);
  }
  float al = acc * scale;
  alpha[idx] = al;
  atomicMaxF(&amax[(size_t)d * heads + h], al);
}

// ---------------------------------------------------------------------------
// pass 2: w = exp(alpha - amax[dst]); denom += w; out[dst] += (v[src]+e)*w
// (normalization by denom is deferred to the node pass -> single edge sweep)
// ---------------------------------------------------------------------------
__global__ void k_attn_agg(const float* __restrict__ v, const float* __restrict__ emb,
                           const float* __restrict__ We, const int* __restrict__ src,
                           const int* __restrict__ dst, const float* __restrict__ alpha,
                           const float* __restrict__ amax, float* __restrict__ denom,
                           float* __restrict__ outb, int heads, int dout, int hc, int tot) {
  int idx = blockIdx.x * blockDim.x + threadIdx.x;
  if (idx >= tot) return;
  int e = idx / heads, h = idx - e * heads;
  int s = src[e], d = dst[e];
  float w = __expf(alpha[idx] - amax[(size_t)d * heads + h]);
  atomicAdd(&denom[(size_t)d * heads + h], w);
  const float* vr   = v + (size_t)s * hc + h * dout;
  const float* em   = emb + (size_t)e * EMB_C;
  const float* Wec  = We + h * dout;
  float*       orow = outb + (size_t)d * hc + h * dout;
  for (int c = 0; c < dout; ++c) {
    float ec = 0.f;
#pragma unroll
    for (int j = 0; j < EMB_C; ++j) ec += em[j] * Wec[(size_t)j * hc + c];
    atomicAdd(&orow[c], (vr[c] + ec) * w);
  }
}

// ---------------------------------------------------------------------------
// pass 3: out /= denom; beta-gated skip; ReLU; write next-layer bf16 input
// ---------------------------------------------------------------------------
__global__ void k_node_update(const float* __restrict__ outb, const float* __restrict__ denom,
                              const float* __restrict__ xr, const float* __restrict__ Wbeta,
                              __bf16* __restrict__ xnext, float* __restrict__ h_out,
                              int heads, int dout, int hc) {
  int n = blockIdx.x * blockDim.x + threadIdx.x;
  if (n >= N_NODES) return;
  const float* orow = outb + (size_t)n * hc;
  const float* xrow = xr + (size_t)n * hc;
  float s = 0.f;
  for (int c = 0; c < hc; ++c) {
    float dn = denom[(size_t)n * heads + c / dout] + 1e-16f;
    float o  = orow[c] / dn;
    float x  = xrow[c];
    // concat([out, x_r, out - x_r]) @ Wbeta, folded
    s += o * (Wbeta[c] + Wbeta[2 * hc + c]) + x * (Wbeta[hc + c] - Wbeta[2 * hc + c]);
  }
  float beta = 1.f / (1.f + __expf(-s));
  for (int c = 0; c < hc; ++c) {
    float dn = denom[(size_t)n * heads + c / dout] + 1e-16f;
    float o  = orow[c] / dn;
    float x  = xrow[c];
    float y  = fmaxf(beta * x + (1.f - beta) * o, 0.f);
    xnext[(size_t)n * hc + c] = (__bf16)y;
    if (h_out) h_out[(size_t)n * hc + c] = y;
  }
}

// ---------------------------------------------------------------------------
// final per-edge MLP: relu([h[s], h[d], E_emb] @ W1 + b1) @ W2 + b2
// ---------------------------------------------------------------------------
__global__ void k_final_mlp(const float* __restrict__ h3, const float* __restrict__ eemb,
                            const int* __restrict__ src, const int* __restrict__ dst,
                            const float* __restrict__ W1, const float* __restrict__ b1,
                            const float* __restrict__ W2, const float* __restrict__ b2,
                            float* __restrict__ out, int ne) {
  int e = blockIdx.x * blockDim.x + threadIdx.x;
  if (e >= ne) return;
  const float* hs = h3 + (size_t)src[e] * 32;
  const float* hd = h3 + (size_t)dst[e] * 32;
  const float* em = eemb + (size_t)e * EMB_C;
  float zs[32], zd[32], ze[EMB_C];
#pragma unroll
  for (int i = 0; i < 32; ++i) { zs[i] = hs[i]; zd[i] = hd[i]; }
#pragma unroll
  for (int i = 0; i < EMB_C; ++i) ze[i] = em[i];
  float acc = b2[0];
  for (int j = 0; j < 32; ++j) {
    float t = b1[j];
#pragma unroll
    for (int i = 0; i < 32; ++i) t += zs[i] * W1[i * 32 + j];
#pragma unroll
    for (int i = 0; i < 32; ++i) t += zd[i] * W1[(32 + i) * 32 + j];
#pragma unroll
    for (int i = 0; i < EMB_C; ++i) t += ze[i] * W1[(64 + i) * 32 + j];
    acc += fmaxf(t, 0.f) * W2[j];
  }
  out[e] = acc;
}

// ---------------------------------------------------------------------------
// host launcher
// ---------------------------------------------------------------------------
extern "C" void kernel_launch(void* const* d_in, const int* in_sizes, int n_in,
                              void* d_out, int out_size, void* d_ws, size_t ws_size,
                              hipStream_t stream) {
  (void)in_sizes; (void)n_in; (void)out_size; (void)ws_size;

  const float* x_dense = (const float*)d_in[0];
  const int*   Aei     = (const int*)d_in[1];
  const float* Aea     = (const float*)d_in[2];
  const int*   Eei     = (const int*)d_in[3];
  const float* Eea     = (const float*)d_in[4];
  const float* Wedge   = (const float*)d_in[5];

  struct LP {
    const float *Wq, *bq, *Wk, *bk, *Wv, *bv, *We, *Wskip, *bskip, *Wbeta;
    int din, dout, heads;
  } lp[3];
  const int dims[3][3] = {{96, 128, 4}, {512, 64, 4}, {256, 32, 1}};
  for (int l = 0; l < 3; ++l) {
    int b = 6 + l * 10;
    lp[l].Wq = (const float*)d_in[b + 0]; lp[l].bq    = (const float*)d_in[b + 1];
    lp[l].Wk = (const float*)d_in[b + 2]; lp[l].bk    = (const float*)d_in[b + 3];
    lp[l].Wv = (const float*)d_in[b + 4]; lp[l].bv    = (const float*)d_in[b + 5];
    lp[l].We = (const float*)d_in[b + 6]; lp[l].Wskip = (const float*)d_in[b + 7];
    lp[l].bskip = (const float*)d_in[b + 8]; lp[l].Wbeta = (const float*)d_in[b + 9];
    lp[l].din = dims[l][0]; lp[l].dout = dims[l][1]; lp[l].heads = dims[l][2];
  }
  const float* mW1 = (const float*)d_in[36];
  const float* mb1 = (const float*)d_in[37];
  const float* mW2 = (const float*)d_in[38];
  const float* mb2 = (const float*)d_in[39];

  // workspace carve-up
  char* p = (char*)d_ws;
  auto carve = [&](size_t bytes) -> void* {
    void* r = (void*)p;
    p += (bytes + 255) & ~(size_t)255;
    return r;
  };
  __bf16* xb    = (__bf16*)carve((size_t)N_NODES * 512 * 2);   // current layer input, bf16
  __bf16* wt    = (__bf16*)carve((size_t)512 * 512 * 2);       // transposed bf16 weight (reused)
  float*  qb    = (float*)carve((size_t)N_NODES * 512 * 4);
  float*  kb    = (float*)carve((size_t)N_NODES * 512 * 4);
  float*  vb    = (float*)carve((size_t)N_NODES * 512 * 4);
  float*  xrb   = (float*)carve((size_t)N_NODES * 512 * 4);
  float*  outb  = (float*)carve((size_t)N_NODES * 512 * 4);
  float*  amax  = (float*)carve((size_t)N_NODES * 4 * 4);
  float*  denom = (float*)carve((size_t)N_NODES * 4 * 4);
  float*  alpha = (float*)carve((size_t)EA_EDGES * 4 * 4);
  float*  Aemb  = (float*)carve((size_t)EA_EDGES * EMB_C * 4);
  float*  Eemb  = (float*)carve((size_t)EE_EDGES * EMB_C * 4);
  float*  h3    = (float*)carve((size_t)N_NODES * 32 * 4);

  const int* srcA = Aei;
  const int* dstA = Aei + EA_EDGES;
  const int* srcE = Eei;
  const int* dstE = Eei + EE_EDGES;

  // stage 0: edge embeddings + input conversion
  k_edge_embed<<<(EA_EDGES + 255) / 256, 256, 0, stream>>>(Aea, Wedge, Aemb, EA_EDGES);
  k_edge_embed<<<(EE_EDGES + 255) / 256, 256, 0, stream>>>(Eea, Wedge, Eemb, EE_EDGES);
  k_f32_to_bf16<<<(N_NODES * IN_DIM_C + 255) / 256, 256, 0, stream>>>(x_dense, xb,
                                                                      N_NODES * IN_DIM_C);

  for (int l = 0; l < 3; ++l) {
    const int din = lp[l].din, dout = lp[l].dout, heads = lp[l].heads;
    const int hc = heads * dout;

    const float* Ws[4] = {lp[l].Wq, lp[l].Wk, lp[l].Wv, lp[l].Wskip};
    const float* bs[4] = {lp[l].bq, lp[l].bk, lp[l].bv, lp[l].bskip};
    float*       Cs[4] = {qb, kb, vb, xrb};
    for (int g = 0; g < 4; ++g) {
      int nel = din * hc;
      k_transpose_to_bf16<<<(nel + 255) / 256, 256, 0, stream>>>(Ws[g], wt, din, hc);
      int tiles_n = hc / 16;
      dim3 grid(N_NODES / 16, (tiles_n + 15) / 16);   // 8 waves x 2 tiles = 16 tiles/block
      k_gemm_bf16_wmma<<<grid, 256, 0, stream>>>(xb, wt, bs[g], Cs[g], din, hc);
    }

    k_fill<<<(N_NODES * heads + 255) / 256, 256, 0, stream>>>(amax, -INFINITY, N_NODES * heads);
    k_fill<<<(N_NODES * heads + 255) / 256, 256, 0, stream>>>(denom, 0.f, N_NODES * heads);
    k_fill<<<(N_NODES * hc + 255) / 256, 256, 0, stream>>>(outb, 0.f, N_NODES * hc);

    const float scale = 1.0f / sqrtf((float)dout);
    const int tot = EA_EDGES * heads;
    k_attn_alpha<<<(tot + 255) / 256, 256, 0, stream>>>(qb, kb, Aemb, lp[l].We, srcA, dstA,
                                                        alpha, amax, heads, dout, hc, tot, scale);
    k_attn_agg<<<(tot + 255) / 256, 256, 0, stream>>>(vb, Aemb, lp[l].We, srcA, dstA, alpha,
                                                      amax, denom, outb, heads, dout, hc, tot);
    k_node_update<<<(N_NODES + 255) / 256, 256, 0, stream>>>(outb, denom, xrb, lp[l].Wbeta, xb,
                                                             (l == 2) ? h3 : (float*)nullptr,
                                                             heads, dout, hc);
  }

  k_final_mlp<<<(EE_EDGES + 255) / 256, 256, 0, stream>>>(h3, Eemb, srcE, dstE, mW1, mb1, mW2,
                                                          mb2, (float*)d_out, EE_EDGES);
}
